// LagrangianSourceIntegration_75058848464945
// MI455X (gfx1250) — matrix operand order repaired and checked by reading, not compiled, not run on hardware
//
#include <hip/hip_runtime.h>
#include <hip/hip_bf16.h>

typedef __attribute__((ext_vector_type(16))) __bf16        v16bf;
typedef __attribute__((ext_vector_type(8)))  float         v8f;
typedef __attribute__((ext_vector_type(4)))  unsigned int  u32x4;

union FragBF { u32x4 q[2]; v16bf v; };

// round-to-nearest-even float -> bf16 bits
__device__ __forceinline__ unsigned short f2bf(float f) {
  union { float f; unsigned int u; } c; c.f = f;
  unsigned int u = c.u + 0x7FFFu + ((c.u >> 16) & 1u);
  return (unsigned short)(u >> 16);
}

// --- ws layout: per-source transposed+padded bf16 weight images (LDS-identical) ---
// W1t[s]:  128 rows x 264 elems  (33792 elems / 67584 B per source)
// W2t[s]:  256 rows x 136 elems  (34816 elems / 69632 B per source)
// Wc1t[s]:  32 rows x 264 elems  ( 8448 elems / 16896 B per source)
constexpr int W1T_ELE  = 128 * 264;
constexpr int W2T_ELE  = 256 * 136;
constexpr int WC1T_ELE = 32 * 264;
constexpr int WS_W2_OFF  = 16 * W1T_ELE;             // 540672
constexpr int WS_WC1_OFF = WS_W2_OFF + 16 * W2T_ELE; // 1097728

// A-matrix fragment (16x32 bf16): lane holds row = lane%16,
// K chunks [half*8 .. +7] and [16+half*8 .. +7]  (two ds_load_b128)
__device__ __forceinline__ v16bf frag_k32_a(const unsigned short* buf, int stride,
                                            int row, int kt, int half) {
  const unsigned short* p = buf + row * stride + kt * 32 + half * 8;
  FragBF f;
  f.q[0] = *(const u32x4*)(p);
  f.q[1] = *(const u32x4*)(p + 16);
  return f.v;
}

// B-matrix fragment (32x16 bf16) from transposed LDS tile Bt[n][k]:
// lane holds col = lane%16, K chunks [half*8 .. +7] and [16+half*8 .. +7]  (two ds_load_b128)
__device__ __forceinline__ v16bf frag_k32_b(const unsigned short* buf, int stride,
                                            int col, int kt, int half) {
  const unsigned short* p = buf + col * stride + kt * 32 + half * 8;
  FragBF f;
  f.q[0] = *(const u32x4*)(p);
  f.q[1] = *(const u32x4*)(p + 16);
  return f.v;
}

// ---------- prep: convert/transpose/pad all weights to bf16 images, once ----------
__global__ __launch_bounds__(256)
void LSI_75058848464945_prep(const float* __restrict__ W1, const float* __restrict__ W2,
                             const float* __restrict__ Wc1, unsigned short* __restrict__ ws)
{
  unsigned short* wsW1  = ws;
  unsigned short* wsW2  = ws + WS_W2_OFF;
  unsigned short* wsWc1 = ws + WS_WC1_OFF;
  const int stride = gridDim.x * blockDim.x;
  const int t0 = blockIdx.x * blockDim.x + threadIdx.x;

  // W1 [16,256,128] -> wsW1[s][n<128][k<256]
  for (int idx = t0; idx < 16 * 128 * 256; idx += stride) {
    int k = idx & 255, n = (idx >> 8) & 127, s = idx >> 15;
    wsW1[s * W1T_ELE + n * 264 + k] = f2bf(W1[(s * 256 + k) * 128 + n]);
  }
  // W2 [16,128,256] -> wsW2[s][n<256][k<128]
  for (int idx = t0; idx < 16 * 256 * 128; idx += stride) {
    int k = idx & 127, n = (idx >> 7) & 255, s = idx >> 15;
    wsW2[s * W2T_ELE + n * 136 + k] = f2bf(W2[(s * 128 + k) * 256 + n]);
  }
  // Wc1 [16,256,32] -> wsWc1[s][n<32][k<256]
  for (int idx = t0; idx < 16 * 32 * 256; idx += stride) {
    int k = idx & 255, n = (idx >> 8) & 31, s = idx >> 13;
    wsWc1[s * WC1T_ELE + n * 264 + k] = f2bf(Wc1[(s * 256 + k) * 32 + n]);
  }
}

// ---------- main fused kernel: 64-row batch tile, 16 waves ----------
__global__ __launch_bounds__(512)
void LagrangianSourceIntegration_75058848464945_kernel(
    const float* __restrict__ x,
    const float* __restrict__ b1,  const float* __restrict__ b2,
    const float* __restrict__ bc1, const float* __restrict__ Wc2,
    const float* __restrict__ bc2, const float* __restrict__ lam,
    const unsigned short* __restrict__ ws, float* __restrict__ out)
{
  constexpr int D = 256, H = 128, S = 16, K = 32;
  constexpr int BT = 64;    // batch rows per block (4 M-tiles of 16)
  constexpr int WS1 = 264;  // row stride (elems) for K=256 rows, +16B pad -> conflict-free b128
  constexpr int WS2 = 136;  // row stride (elems) for K=128 rows

  __shared__ __align__(16) unsigned short wbuf[256 * WS2]; // 69632B: fits all three weight images
  __shared__ __align__(16) unsigned short xA[BT * WS1];    // x tile bf16      (33792B)
  __shared__ __align__(16) unsigned short hbuf[BT * WS2];  // h tile bf16      (17408B)
  __shared__ __align__(16) unsigned short obuf[BT * WS1];  // outs tile bf16   (33792B)
  __shared__ float chbuf[BT * 33];                         // ch tile f32      ( 8448B)
  __shared__ float cwbuf[BT];
  __shared__ float cwsum[BT];

  const int tid  = threadIdx.x;
  const int lane = tid & 31;
  const int wv   = tid >> 5;        // 0..15
  const int mt   = wv >> 2;         // M-tile (0..3) this wave group owns
  const int wq   = wv & 3;          // wave index within its M-group (0..3)
  const int lrow = lane & 15;       // row (A/D) or col (B/D) inside 16-tile
  const int half = lane >> 4;
  const int rowBase = blockIdx.x * BT;
  const int arow = mt * 16 + lrow;  // A-fragment row inside block tile

  const unsigned short* wsW1  = ws;
  const unsigned short* wsW2  = ws + WS_W2_OFF;
  const unsigned short* wsWc1 = ws + WS_WC1_OFF;

  // Stage x tile -> bf16 LDS (coalesced global reads)
  for (int idx = tid; idx < BT * D; idx += 512) {
    int r = idx >> 8, c = idx & 255;
    xA[r * WS1 + c] = f2bf(x[(size_t)(rowBase + r) * D + c]);
  }
  if (tid < BT) cwsum[tid] = 0.0f;

  v8f accOut[4] = {};   // persistent weighted-sum fragments (this wave's 64 columns of its M-tile)
  __syncthreads();

  for (int s = 0; s < S; ++s) {
    // ---- stage W1[s]^T image: pure b128 copy ----
    {
      const u32x4* src = (const u32x4*)(wsW1 + (size_t)s * W1T_ELE);
      u32x4* dst = (u32x4*)wbuf;
      for (int i = tid; i < W1T_ELE / 8; i += 512) dst[i] = src[i];
    }
    __syncthreads();

    // ---- GEMM1: h[64,128] = relu(x @ W1 + b1); 2 N-tiles per wave ----
    #pragma unroll
    for (int t = 0; t < 2; ++t) {
      const int nt = wq * 2 + t;
      v8f acc = {};
      #pragma unroll
      for (int kt = 0; kt < 8; ++kt) {
        v16bf a = frag_k32_a(xA,   WS1, arow,           kt, half);
        v16bf b = frag_k32_b(wbuf, WS1, nt * 16 + lrow, kt, half);
        acc = __builtin_amdgcn_wmma_f32_16x16x32_bf16(false, a, false, b,
                                                      (short)0, acc, false, false);
      }
      const int n = nt * 16 + lrow;
      const float bias = b1[s * H + n];
      #pragma unroll
      for (int r = 0; r < 8; ++r) {
        float v = acc[r] + bias;
        hbuf[(mt * 16 + r + 8 * half) * WS2 + n] = f2bf(v > 0.0f ? v : 0.0f);
      }
    }
    __syncthreads();

    // ---- stage W2[s]^T image ----
    {
      const u32x4* src = (const u32x4*)(wsW2 + (size_t)s * W2T_ELE);
      u32x4* dst = (u32x4*)wbuf;
      for (int i = tid; i < W2T_ELE / 8; i += 512) dst[i] = src[i];
    }
    __syncthreads();

    // ---- GEMM2: outs[64,256] = h @ W2 + b2; 4 N-tiles per wave, kept in regs ----
    v8f outv[4];
    #pragma unroll
    for (int t = 0; t < 4; ++t) {
      const int nt = wq * 4 + t;
      v8f acc = {};
      #pragma unroll
      for (int kt = 0; kt < 4; ++kt) {
        v16bf a = frag_k32_a(hbuf, WS2, arow,           kt, half);
        v16bf b = frag_k32_b(wbuf, WS2, nt * 16 + lrow, kt, half);
        acc = __builtin_amdgcn_wmma_f32_16x16x32_bf16(false, a, false, b,
                                                      (short)0, acc, false, false);
      }
      const int n = nt * 16 + lrow;
      const float bias = b2[s * D + n];
      #pragma unroll
      for (int r = 0; r < 8; ++r) {
        acc[r] += bias;
        obuf[(mt * 16 + r + 8 * half) * WS1 + n] = f2bf(acc[r]);
      }
      outv[t] = acc;
    }
    __syncthreads();

    // ---- stage Wc1[s]^T image ----
    {
      const u32x4* src = (const u32x4*)(wsWc1 + (size_t)s * WC1T_ELE);
      u32x4* dst = (u32x4*)wbuf;
      for (int i = tid; i < WC1T_ELE / 8; i += 512) dst[i] = src[i];
    }
    __syncthreads();

    // ---- GEMM3: ch[64,32] = relu(outs @ Wc1 + bc1); 2 waves per M-group ----
    if (wq < 2) {
      v8f acc = {};
      #pragma unroll
      for (int kt = 0; kt < 8; ++kt) {
        v16bf a = frag_k32_a(obuf, WS1, arow,           kt, half);
        v16bf b = frag_k32_b(wbuf, WS1, wq * 16 + lrow, kt, half);
        acc = __builtin_amdgcn_wmma_f32_16x16x32_bf16(false, a, false, b,
                                                      (short)0, acc, false, false);
      }
      const int n = wq * 16 + lrow;
      const float bias = bc1[s * K + n];
      #pragma unroll
      for (int r = 0; r < 8; ++r) {
        float v = acc[r] + bias;
        chbuf[(mt * 16 + r + 8 * half) * 33 + n] = v > 0.0f ? v : 0.0f;
      }
    }
    __syncthreads();

    // ---- conf = sigmoid(ch . Wc2 + bc2); cw = lam*conf; one thread per row ----
    if (tid < BT) {
      float acc = bc2[s];
      #pragma unroll
      for (int k = 0; k < K; ++k) acc += chbuf[tid * 33 + k] * Wc2[s * K + k];
      float cw = lam[s] * (1.0f / (1.0f + __expf(-acc)));
      cwbuf[tid] = cw;
      cwsum[tid] += cw;
    }
    __syncthreads();

    // ---- accumulate cw[row] * outs into persistent fragments ----
    #pragma unroll
    for (int t = 0; t < 4; ++t) {
      #pragma unroll
      for (int r = 0; r < 8; ++r) {
        accOut[t][r] += cwbuf[mt * 16 + r + 8 * half] * outv[t][r];
      }
    }
    __syncthreads();
  }

  // ---- normalize and store: out = accOut / (sum_cw + 1e-6) ----
  #pragma unroll
  for (int t = 0; t < 4; ++t) {
    const int n = (wq * 4 + t) * 16 + lrow;
    #pragma unroll
    for (int r = 0; r < 8; ++r) {
      const int m = mt * 16 + r + 8 * half;
      out[(size_t)(rowBase + m) * D + n] = accOut[t][r] / (cwsum[m] + 1e-6f);
    }
  }
}

extern "C" void kernel_launch(void* const* d_in, const int* in_sizes, int n_in,
                              void* d_out, int out_size, void* d_ws, size_t ws_size,
                              hipStream_t stream) {
  const float* x   = (const float*)d_in[0];
  const float* W1  = (const float*)d_in[1];
  const float* b1  = (const float*)d_in[2];
  const float* W2  = (const float*)d_in[3];
  const float* b2  = (const float*)d_in[4];
  const float* Wc1 = (const float*)d_in[5];
  const float* bc1 = (const float*)d_in[6];
  const float* Wc2 = (const float*)d_in[7];
  const float* bc2 = (const float*)d_in[8];
  const float* lam = (const float*)d_in[9];
  float* out = (float*)d_out;
  unsigned short* ws = (unsigned short*)d_ws;

  // 1) one-shot weight convert/transpose/pad into d_ws (bf16 LDS-image layout)
  hipLaunchKernelGGL(LSI_75058848464945_prep, dim3(512), dim3(256), 0, stream,
                     W1, W2, Wc1, ws);

  // 2) fused per-tile kernel: all GEMMs on bf16 WMMA, fp32 accumulate
  const int B = 16384;
  hipLaunchKernelGGL(LagrangianSourceIntegration_75058848464945_kernel,
                     dim3(B / 64), dim3(512), 0, stream,
                     x, b1, b2, bc1, Wc2, bc2, lam, ws, out);
}
